// MoETransformerDecoderLayerBase_70720931496634
// MI455X (gfx1250) — matrix operand order repaired and checked by reading, not compiled, not run on hardware
//
#include <hip/hip_runtime.h>

typedef _Float16 h8v  __attribute__((ext_vector_type(8)));
typedef _Float16 v16h __attribute__((ext_vector_type(16)));
typedef float    v8f  __attribute__((ext_vector_type(8)));

#define SHUF16(lo, hi) __builtin_shufflevector(lo, hi, 0,1,2,3,4,5,6,7,8,9,10,11,12,13,14,15)

static __device__ __forceinline__ v8f wmma16(v16h a, v16h b, v8f c) {
  return __builtin_amdgcn_wmma_f32_16x16x32_f16(false, a, false, b, (short)0, c, false, false);
}

// ---- gfx1250 async global->LDS path (ASYNCcnt), guarded with fallback ------
#if defined(__has_builtin)
#if __has_builtin(__builtin_amdgcn_global_load_async_to_lds_b128)
#define HAS_ASYNC_LDS 1
#endif
#endif
#ifndef HAS_ASYNC_LDS
#define HAS_ASYNC_LDS 0
#endif

typedef int v4i_gcc __attribute__((vector_size(16)));
typedef __attribute__((address_space(1))) v4i_gcc as1_v4i;
typedef __attribute__((address_space(3))) v4i_gcc as3_v4i;

static __device__ __forceinline__ void async_copy16(const _Float16* g, _Float16* l) {
#if HAS_ASYNC_LDS
  __builtin_amdgcn_global_load_async_to_lds_b128((as1_v4i*)g, (as3_v4i*)l, 0, 0);
#else
  (void)g; (void)l;
#endif
}

template<int N>
static __device__ __forceinline__ void wait_async() {
#if defined(__has_builtin)
#if __has_builtin(__builtin_amdgcn_s_wait_asynccnt)
#define HAS_WAIT_ASYNC 1
  __builtin_amdgcn_s_wait_asynccnt(N);
#endif
#endif
#ifndef HAS_WAIT_ASYNC
  asm volatile("s_wait_asynccnt %0" :: "i"(N) : "memory");
#endif
}

// ---------------------------------------------------------------------------
// Cast fp32 [R][C] -> f16 transposed [C][R], batched over blockIdx.z
// ---------------------------------------------------------------------------
__global__ __launch_bounds__(256)
void cast_transpose_kernel(const float* __restrict__ in, _Float16* __restrict__ out,
                           int R, int C)
{
  __shared__ float tile[32][33];
  size_t bofs = (size_t)blockIdx.z * R * C;
  in += bofs; out += bofs;
  int tid = threadIdx.x, tx = tid & 31, ty = tid >> 5;
  int c0 = blockIdx.x * 32, r0 = blockIdx.y * 32;
#pragma unroll
  for (int j = 0; j < 4; j++)
    tile[ty + j * 8][tx] = in[(size_t)(r0 + ty + j * 8) * C + c0 + tx];
  __syncthreads();
#pragma unroll
  for (int j = 0; j < 4; j++)
    out[(size_t)(c0 + ty + j * 8) * R + r0 + tx] = (_Float16)tile[tx][ty + j * 8];
}

// ---------------------------------------------------------------------------
// LayerNorm row (D=1024) -> f16
// ---------------------------------------------------------------------------
__global__ __launch_bounds__(256)
void ln_kernel(const float* __restrict__ x, const float* __restrict__ g,
               const float* __restrict__ bb, _Float16* __restrict__ out)
{
  int row = blockIdx.x;
  const float* xr = x + (size_t)row * 1024;
  int tid = threadIdx.x;
  __shared__ float r1[256], r2[256];
  float v[4]; float s = 0.f, sq = 0.f;
#pragma unroll
  for (int i = 0; i < 4; i++) { v[i] = xr[tid + i * 256]; s += v[i]; sq += v[i] * v[i]; }
  r1[tid] = s; r2[tid] = sq; __syncthreads();
  for (int o = 128; o > 0; o >>= 1) {
    if (tid < o) { r1[tid] += r1[tid + o]; r2[tid] += r2[tid + o]; }
    __syncthreads();
  }
  float mean = r1[0] * (1.f / 1024.f);
  float var = r2[0] * (1.f / 1024.f) - mean * mean;
  float rstd = rsqrtf(var + 1e-5f);
#pragma unroll
  for (int i = 0; i < 4; i++) {
    int j = tid + i * 256;
    out[(size_t)row * 1024 + j] = (_Float16)((v[i] - mean) * rstd * g[j] + bb[j]);
  }
}

// ---------------------------------------------------------------------------
// Generic f16 WMMA GEMM: C[M,N] = A[M,K] @ B[K,N] (+bias, +epilogue)
// BT is B transposed: [N][K] f16. Block tile 128x64, 8 waves of 32x32.
// Double-buffered LDS staging via GLOBAL_LOAD_ASYNC_TO_LDS_B128 (ASYNCcnt).
// MODE 0: OutH f16 = (acc+bias)*scale                  (QKV projections)
// MODE 1: OutF f32 = acc+bias+Res0[row]                (O-projection + residual)
// MODE 2: A rows gathered via perm, OutH f16 = relu    (expert FFN1)
// MODE 3: A slot rows, scatter f32 = acc+bias+xs+Res0  (expert FFN2 + residuals)
// ---------------------------------------------------------------------------
template<int MODE>
__global__ __launch_bounds__(256)
void gemm_kernel(const _Float16* __restrict__ A, const _Float16* __restrict__ BT,
                 const float* __restrict__ bias,
                 _Float16* __restrict__ OutH, float* __restrict__ OutF,
                 const float* __restrict__ Res0, const float* __restrict__ xsTok,
                 const int* __restrict__ perm, const int* __restrict__ offs,
                 const int* __restrict__ cnts,
                 int M, int N, int K, float scale)
{
  int cnt = M, slot0 = 0;
  if (MODE >= 2) {
    int e = blockIdx.z;
    cnt = cnts[e];
    slot0 = offs[e];
    BT += (size_t)e * N * K;
    bias += (size_t)e * N;
  }
  int m0 = blockIdx.y * 128;
  if (m0 >= cnt) return;
  int n0 = blockIdx.x * 64;

  __shared__ __align__(16) _Float16 As[2][128][40];
  __shared__ __align__(16) _Float16 Bs[2][64][40];

  int tid = threadIdx.x;
  int lane = tid & 31, w = tid >> 5;
  int wr = w >> 1, wc = w & 1;
  int ln = lane & 15, hl = lane >> 4;

  // A-row for staging: clamp out-of-range rows to a valid one (their outputs
  // are never written), keeping issue EXEC-uniform.
  int arow = tid >> 1, ahalf = tid & 1;
  int lr = m0 + arow;
  if (lr >= cnt) lr = cnt - 1;
  const _Float16* aptr;
  if (MODE == 2)      aptr = A + (size_t)perm[slot0 + lr] * K;
  else if (MODE == 3) aptr = A + (size_t)(slot0 + lr) * K;
  else                aptr = A + (size_t)lr * K;
  aptr += ahalf * 16;

  int brow = tid >> 2, bseg = tid & 3;
  const _Float16* bptr = BT + (size_t)(n0 + brow) * K + bseg * 8;

  v8f acc[2][2] = {};

#if HAS_ASYNC_LDS
  async_copy16(aptr,     &As[0][arow][ahalf * 16]);
  async_copy16(aptr + 8, &As[0][arow][ahalf * 16 + 8]);
  async_copy16(bptr,     &Bs[0][brow][bseg * 8]);
#endif

  int kt = 0;
  for (int k0 = 0; k0 < K; k0 += 32, kt ^= 1) {
#if HAS_ASYNC_LDS
    if (k0 + 32 < K) {
      int nb = kt ^ 1;
      async_copy16(aptr + k0 + 32,     &As[nb][arow][ahalf * 16]);
      async_copy16(aptr + k0 + 40,     &As[nb][arow][ahalf * 16 + 8]);
      async_copy16(bptr + k0 + 32,     &Bs[nb][brow][bseg * 8]);
      wait_async<3>();   // the 3 older (current-tile) async ops are done
    } else {
      wait_async<0>();
    }
    __syncthreads();
#else
    {
      h8v a0 = *(const h8v*)(aptr + k0);
      h8v a1 = *(const h8v*)(aptr + k0 + 8);
      h8v bvv = *(const h8v*)(bptr + k0);
      *(h8v*)&As[kt][arow][ahalf * 16] = a0;
      *(h8v*)&As[kt][arow][ahalf * 16 + 8] = a1;
      *(h8v*)&Bs[kt][brow][bseg * 8] = bvv;
      __syncthreads();
    }
#endif

    v16h fa[2], fb[2];
#pragma unroll
    for (int i = 0; i < 2; i++) {
      const _Float16* p = &As[kt][wr * 32 + i * 16 + ln][0];
      h8v lo = *(const h8v*)(p + hl * 8);
      h8v hi = *(const h8v*)(p + 16 + hl * 8);
      fa[i] = SHUF16(lo, hi);
    }
#pragma unroll
    for (int i = 0; i < 2; i++) {
      const _Float16* p = &Bs[kt][wc * 32 + i * 16 + ln][0];
      h8v lo = *(const h8v*)(p + hl * 8);
      h8v hi = *(const h8v*)(p + 16 + hl * 8);
      fb[i] = SHUF16(lo, hi);
    }
#pragma unroll
    for (int i = 0; i < 2; i++)
#pragma unroll
      for (int j = 0; j < 2; j++)
        acc[i][j] = wmma16(fa[i], fb[j], acc[i][j]);
    __syncthreads();   // all waves done reading buf kt before it is refilled
  }

#pragma unroll
  for (int i = 0; i < 2; i++) {
#pragma unroll
    for (int j = 0; j < 2; j++) {
      int col = n0 + wc * 32 + j * 16 + ln;
      float bval = bias[col];
#pragma unroll
      for (int v = 0; v < 8; v++) {
        int mrow = wr * 32 + i * 16 + hl * 8 + v;
        if (m0 + mrow >= cnt) continue;
        float val = acc[i][j][v] + bval;
        if (MODE == 0) {
          OutH[(size_t)(m0 + mrow) * N + col] = (_Float16)(val * scale);
        } else if (MODE == 1) {
          size_t o = (size_t)(m0 + mrow) * N + col;
          OutF[o] = val + Res0[o];
        } else if (MODE == 2) {
          OutH[(size_t)(slot0 + m0 + mrow) * N + col] = (_Float16)fmaxf(val, 0.f);
        } else {
          int tok = perm[slot0 + m0 + mrow];
          int orow = (tok & 2047) * 2 + (tok >> 11);   // t*B + b
          OutF[(size_t)orow * N + col] =
              val + xsTok[(size_t)tok * N + col] + Res0[(size_t)orow * N + col];
        }
      }
    }
  }
}

// ---------------------------------------------------------------------------
// Flash attention, one wave per block. Each wave: 16 query rows of one (b,h).
// Scores/PV via v_wmma_f32_16x16x32_f16, online softmax with 16-lane shuffles.
// ---------------------------------------------------------------------------
__global__ __launch_bounds__(32)
void flash_kernel(const _Float16* __restrict__ qf, const _Float16* __restrict__ kf,
                  const _Float16* __restrict__ vf, _Float16* __restrict__ af)
{
  constexpr int T = 2048, Bc = 2, H = 16, HD = 64, D = 1024;
  int t0 = blockIdx.x * 16;
  int bh = blockIdx.y;
  int b = bh / H, h = bh % H;
  int lane = threadIdx.x;
  int ln = lane & 15, hl = lane >> 4;

  __shared__ __align__(16) _Float16 Pl[16][40];
  __shared__ __align__(16) _Float16 Vt[64][40];

  v16h qA[2];
  {
    int t = t0 + ln;
    const _Float16* qrow = qf + (size_t)(t * Bc + b) * D + h * HD;
#pragma unroll
    for (int j = 0; j < 2; j++) {
      h8v lo = *(const h8v*)(qrow + j * 32 + hl * 8);
      h8v hi = *(const h8v*)(qrow + j * 32 + 16 + hl * 8);
      qA[j] = SHUF16(lo, hi);
    }
  }

  float rM[8], rL[8];
#pragma unroll
  for (int v = 0; v < 8; v++) { rM[v] = -1e30f; rL[v] = 0.f; }
  v8f O[4] = {};

  int send = t0 + 16;                 // causal: keys < t0+16 can contribute
  for (int s0 = 0; s0 < send; s0 += 32) {
    v8f S[2] = {};
#pragma unroll
    for (int kc = 0; kc < 2; kc++) {
      int s = s0 + kc * 16 + ln;
      const _Float16* krow = kf + (size_t)(s * Bc + b) * D + h * HD;
#pragma unroll
      for (int j = 0; j < 2; j++) {
        h8v lo = *(const h8v*)(krow + j * 32 + hl * 8);
        h8v hi = *(const h8v*)(krow + j * 32 + 16 + hl * 8);
        v16h kB = SHUF16(lo, hi);
        S[kc] = wmma16(qA[j], kB, S[kc]);
      }
    }
    // causal mask
#pragma unroll
    for (int kc = 0; kc < 2; kc++) {
      int s = s0 + kc * 16 + ln;
#pragma unroll
      for (int v = 0; v < 8; v++) {
        int t = t0 + hl * 8 + v;
        if (s > t) S[kc][v] = -1e30f;
      }
    }
    // online softmax update
#pragma unroll
    for (int v = 0; v < 8; v++) {
      float mx = fmaxf(S[0][v], S[1][v]);
#pragma unroll
      for (int o = 8; o > 0; o >>= 1) mx = fmaxf(mx, __shfl_xor(mx, o, 16));
      float nM = fmaxf(rM[v], mx);
      float sc = __expf(rM[v] - nM);
      float p0 = __expf(S[0][v] - nM);
      float p1 = __expf(S[1][v] - nM);
      S[0][v] = p0; S[1][v] = p1;
      float rs = p0 + p1;
#pragma unroll
      for (int o = 8; o > 0; o >>= 1) rs += __shfl_xor(rs, o, 16);
      rL[v] = rL[v] * sc + rs;
      rM[v] = nM;
#pragma unroll
      for (int j = 0; j < 4; j++) O[j][v] *= sc;
    }
    __syncthreads();   // prior-iteration LDS reads done
    // stage P (C-layout -> A-layout via LDS) and V^T
#pragma unroll
    for (int kc = 0; kc < 2; kc++)
#pragma unroll
      for (int v = 0; v < 8; v++)
        Pl[hl * 8 + v][kc * 16 + ln] = (_Float16)S[kc][v];
    for (int idx = lane; idx < 32 * 64; idx += 32) {
      int s = idx >> 6, d = idx & 63;
      Vt[d][s] = vf[(size_t)((s0 + s) * Bc + b) * D + h * HD + d];
    }
    __syncthreads();

    v16h pA;
    {
      const _Float16* p = &Pl[ln][0];
      h8v lo = *(const h8v*)(p + hl * 8);
      h8v hi = *(const h8v*)(p + 16 + hl * 8);
      pA = SHUF16(lo, hi);
    }
#pragma unroll
    for (int j = 0; j < 4; j++) {
      const _Float16* p = &Vt[j * 16 + ln][0];
      h8v lo = *(const h8v*)(p + hl * 8);
      h8v hi = *(const h8v*)(p + 16 + hl * 8);
      v16h vB = SHUF16(lo, hi);
      O[j] = wmma16(pA, vB, O[j]);
    }
  }

#pragma unroll
  for (int j = 0; j < 4; j++)
#pragma unroll
    for (int v = 0; v < 8; v++) {
      int t = t0 + hl * 8 + v;
      af[(size_t)(t * Bc + b) * D + h * HD + j * 16 + ln] = (_Float16)(O[j][v] / rL[v]);
    }
}

// ---------------------------------------------------------------------------
// Routing: LN2 -> xs (fp32), gating logits -> argmax -> assign/count,
// expert-LN of xs -> xe (f16)
// ---------------------------------------------------------------------------
__global__ __launch_bounds__(256)
void route_kernel(const float* __restrict__ x2, const float* __restrict__ g2,
                  const float* __restrict__ b2, const float* __restrict__ cent,
                  const float* __restrict__ elng, const float* __restrict__ elnb,
                  float* __restrict__ xs, _Float16* __restrict__ xe,
                  int* __restrict__ assign, int* __restrict__ counts)
{
  int n = blockIdx.x;
  int t = n & 2047, b = n >> 11;              // n = b*T + t
  const float* xr = x2 + (size_t)(t * 2 + b) * 1024;
  int tid = threadIdx.x;
  __shared__ float r1[256], r2[256];
  __shared__ float lg[8];
  __shared__ int se_sh;

  float v[4]; float s = 0.f, sq = 0.f;
#pragma unroll
  for (int i = 0; i < 4; i++) { v[i] = xr[tid + i * 256]; s += v[i]; sq += v[i] * v[i]; }
  r1[tid] = s; r2[tid] = sq; __syncthreads();
  for (int o = 128; o > 0; o >>= 1) {
    if (tid < o) { r1[tid] += r1[tid + o]; r2[tid] += r2[tid + o]; }
    __syncthreads();
  }
  float mean = r1[0] * (1.f / 1024.f);
  float var = r2[0] * (1.f / 1024.f) - mean * mean;
  float rstd = rsqrtf(var + 1e-5f);
  float xv[4];
#pragma unroll
  for (int i = 0; i < 4; i++) {
    int j = tid + i * 256;
    xv[i] = (v[i] - mean) * rstd * g2[j] + b2[j];
    xs[(size_t)n * 1024 + j] = xv[i];
  }
  __syncthreads();
  // gating logits (fp32)
  for (int e = 0; e < 8; e++) {
    float p = 0.f;
#pragma unroll
    for (int i = 0; i < 4; i++) p += xv[i] * cent[(size_t)e * 1024 + tid + i * 256];
    r1[tid] = p; __syncthreads();
    for (int o = 128; o > 0; o >>= 1) {
      if (tid < o) r1[tid] += r1[tid + o];
      __syncthreads();
    }
    if (tid == 0) lg[e] = r1[0];
    __syncthreads();
  }
  if (tid == 0) {
    int be = 0; float bvv = lg[0];
    for (int e = 1; e < 8; e++) if (lg[e] > bvv) { bvv = lg[e]; be = e; }
    se_sh = be;
    assign[n] = be;
    atomicAdd(&counts[be], 1);
  }
  __syncthreads();
  int e = se_sh;
  // expert LayerNorm of xs
  s = 0.f; sq = 0.f;
#pragma unroll
  for (int i = 0; i < 4; i++) { s += xv[i]; sq += xv[i] * xv[i]; }
  r1[tid] = s; r2[tid] = sq; __syncthreads();
  for (int o = 128; o > 0; o >>= 1) {
    if (tid < o) { r1[tid] += r1[tid + o]; r2[tid] += r2[tid + o]; }
    __syncthreads();
  }
  float m2 = r1[0] * (1.f / 1024.f);
  float v2 = r2[0] * (1.f / 1024.f) - m2 * m2;
  float rs2 = rsqrtf(v2 + 1e-5f);
#pragma unroll
  for (int i = 0; i < 4; i++) {
    int j = tid + i * 256;
    xe[(size_t)n * 1024 + j] =
        (_Float16)((xv[i] - m2) * rs2 * elng[(size_t)e * 1024 + j] + elnb[(size_t)e * 1024 + j]);
  }
}

__global__ void init_kernel(int* p) { if (threadIdx.x < 16) p[threadIdx.x] = 0; }

__global__ void scan_kernel(const int* __restrict__ counts, int* __restrict__ offs)
{
  if (threadIdx.x == 0) {
    int a = 0;
    for (int e = 0; e < 8; e++) { offs[e] = a; a += counts[e]; }
    offs[8] = a;
  }
}

__global__ __launch_bounds__(256)
void scatter_kernel(const int* __restrict__ assign, const int* __restrict__ offs,
                    int* __restrict__ cursor, int* __restrict__ perm)
{
  int n = blockIdx.x * 256 + threadIdx.x;
  int e = assign[n];
  int slot = offs[e] + atomicAdd(&cursor[e], 1);
  perm[slot] = n;
}

// ---------------------------------------------------------------------------
extern "C" void kernel_launch(void* const* d_in, const int* in_sizes, int n_in,
                              void* d_out, int out_size, void* d_ws, size_t ws_size,
                              hipStream_t stream)
{
  (void)in_sizes; (void)n_in; (void)out_size; (void)ws_size;
  constexpr int T = 2048, Bc = 2, D = 1024, H = 16, F = 4096, E = 8, TB = T * Bc;

  const float* x    = (const float*)d_in[0];
  const float* Wq   = (const float*)d_in[1];
  const float* bq   = (const float*)d_in[2];
  const float* Wk   = (const float*)d_in[3];
  const float* bk   = (const float*)d_in[4];
  const float* Wv   = (const float*)d_in[5];
  const float* bv   = (const float*)d_in[6];
  const float* Wo   = (const float*)d_in[7];
  const float* bo   = (const float*)d_in[8];
  const float* ln1g = (const float*)d_in[9];
  const float* ln1b = (const float*)d_in[10];
  const float* ln2g = (const float*)d_in[11];
  const float* ln2b = (const float*)d_in[12];
  const float* cent = (const float*)d_in[13];
  const float* elng = (const float*)d_in[14];
  const float* elnb = (const float*)d_in[15];
  const float* eW1  = (const float*)d_in[16];
  const float* eb1  = (const float*)d_in[17];
  const float* eW2  = (const float*)d_in[18];
  const float* eb2  = (const float*)d_in[19];
  float* out = (float*)d_out;

  char* base = (char*)d_ws;
  size_t off = 0;
  auto alloc = [&](size_t bytes) -> void* {
    void* p = base + off;
    off = (off + bytes + 255) & ~(size_t)255;
    return p;
  };
  _Float16* wqt = (_Float16*)alloc((size_t)D * D * 2);
  _Float16* wkt = (_Float16*)alloc((size_t)D * D * 2);
  _Float16* wvt = (_Float16*)alloc((size_t)D * D * 2);
  _Float16* wot = (_Float16*)alloc((size_t)D * D * 2);
  _Float16* w1t = (_Float16*)alloc((size_t)E * D * F * 2);
  _Float16* w2t = (_Float16*)alloc((size_t)E * D * F * 2);
  _Float16* hfp = (_Float16*)alloc((size_t)TB * D * 2);
  _Float16* qfb = (_Float16*)alloc((size_t)TB * D * 2);
  _Float16* kfb = (_Float16*)alloc((size_t)TB * D * 2);
  _Float16* vfb = (_Float16*)alloc((size_t)TB * D * 2);
  _Float16* afb = (_Float16*)alloc((size_t)TB * D * 2);
  float*    x2  = (float*)alloc((size_t)TB * D * 4);
  float*    xsb = (float*)alloc((size_t)TB * D * 4);
  _Float16* xeb = (_Float16*)alloc((size_t)TB * D * 2);
  _Float16* hh  = (_Float16*)alloc((size_t)TB * F * 2);
  int* ctrl   = (int*)alloc(64 * 4);
  int* counts = ctrl;
  int* cursor = ctrl + 8;
  int* offs   = ctrl + 16;
  int* assign = (int*)alloc((size_t)TB * 4);
  int* perm   = (int*)alloc((size_t)TB * 4);

  dim3 blk(256);

  // weights -> f16 transposed [N][K]
  cast_transpose_kernel<<<dim3(D / 32, D / 32, 1), blk, 0, stream>>>(Wq, wqt, D, D);
  cast_transpose_kernel<<<dim3(D / 32, D / 32, 1), blk, 0, stream>>>(Wk, wkt, D, D);
  cast_transpose_kernel<<<dim3(D / 32, D / 32, 1), blk, 0, stream>>>(Wv, wvt, D, D);
  cast_transpose_kernel<<<dim3(D / 32, D / 32, 1), blk, 0, stream>>>(Wo, wot, D, D);
  cast_transpose_kernel<<<dim3(F / 32, D / 32, E), blk, 0, stream>>>(eW1, w1t, D, F);
  cast_transpose_kernel<<<dim3(D / 32, F / 32, E), blk, 0, stream>>>(eW2, w2t, F, D);

  // attention block
  ln_kernel<<<TB, blk, 0, stream>>>(x, ln1g, ln1b, hfp);
  gemm_kernel<0><<<dim3(D / 64, TB / 128, 1), blk, 0, stream>>>(
      hfp, wqt, bq, qfb, nullptr, nullptr, nullptr, nullptr, nullptr, nullptr,
      TB, D, D, 0.125f);                                   // HD^-0.5 = 1/8
  gemm_kernel<0><<<dim3(D / 64, TB / 128, 1), blk, 0, stream>>>(
      hfp, wkt, bk, kfb, nullptr, nullptr, nullptr, nullptr, nullptr, nullptr,
      TB, D, D, 1.0f);
  gemm_kernel<0><<<dim3(D / 64, TB / 128, 1), blk, 0, stream>>>(
      hfp, wvt, bv, vfb, nullptr, nullptr, nullptr, nullptr, nullptr, nullptr,
      TB, D, D, 1.0f);
  flash_kernel<<<dim3(T / 16, Bc * H), dim3(32), 0, stream>>>(qfb, kfb, vfb, afb);
  gemm_kernel<1><<<dim3(D / 64, TB / 128, 1), blk, 0, stream>>>(
      afb, wot, bo, nullptr, x2, x, nullptr, nullptr, nullptr, nullptr,
      TB, D, D, 1.0f);

  // MoE routing
  init_kernel<<<1, 32, 0, stream>>>(ctrl);
  route_kernel<<<TB, blk, 0, stream>>>(x2, ln2g, ln2b, cent, elng, elnb,
                                       xsb, xeb, assign, counts);
  scan_kernel<<<1, 32, 0, stream>>>(counts, offs);
  scatter_kernel<<<TB / 256, blk, 0, stream>>>(assign, offs, cursor, perm);

  // expert FFN (routed, compacted per-expert rows)
  gemm_kernel<2><<<dim3(F / 64, TB / 128, E), blk, 0, stream>>>(
      xeb, w1t, eb1, hh, nullptr, nullptr, nullptr, perm, offs, counts,
      TB, F, D, 1.0f);
  gemm_kernel<3><<<dim3(D / 64, TB / 128, E), blk, 0, stream>>>(
      hh, w2t, eb2, nullptr, out, x2, xsb, perm, offs, counts,
      TB, D, F, 1.0f);
}
